// Block_26113401159741
// MI455X (gfx1250) — matrix operand order repaired and checked by reading, not compiled, not run on hardware
//
#include <hip/hip_runtime.h>
#include <math.h>
#include <stdint.h>

#define DEV static __device__ __forceinline__

typedef __attribute__((ext_vector_type(16))) __bf16 bf16x16;
typedef __attribute__((ext_vector_type(8)))  __bf16 bf16x8;
typedef __attribute__((ext_vector_type(8)))  float  f32x8;

namespace {
constexpr int B_   = 16;
constexpr int SEQ  = 1025;
constexpr int C_   = 384;
constexpr int NH   = 6;
constexpr int HD   = 64;
constexpr int NP   = 1056;          // padded sequence length (multiple of 32)
constexpr int PIX  = 1024;          // 32x32 spatial tokens
constexpr int HID  = 1536;
constexpr int C4   = 96;
constexpr int MROWS = B_ * SEQ;     // 16400
constexpr int MTOK  = B_ * PIX;     // 16384

constexpr int LDA  = 40;            // padded LDS row (bf16) -> bank-conflict-free
constexpr int ABUF = 64 * LDA;      // one A stage buffer (elems)
constexpr int BBUF = 128 * LDA;     // one B stage buffer (elems)
}

// ---------------------------------------------------------------- WMMA utils

DEV f32x8 wmma_bf16(bf16x16 a, bf16x16 b, f32x8 c) {
  // (neg_a, A, neg_b, B, c_mod, C, reuse_a, reuse_b)
  return __builtin_amdgcn_wmma_f32_16x16x32_bf16(false, a, false, b, (short)0, c,
                                                 false, false);
}

// A fragment (16x32 bf16): lane holds row (lane&15); K chunks at
// base=(lane>>4)*8 : [base..base+7] and [base+16..base+23].
DEV bf16x16 make_a(const __bf16* p) {
  bf16x8 c0 = *(const bf16x8*)p;
  bf16x8 c1 = *(const bf16x8*)(p + 16);
  bf16x16 r;
#pragma unroll
  for (int i = 0; i < 8; ++i) { r[i] = c0[i]; r[i + 8] = c1[i]; }
  return r;
}

// B fragment (32x16 bf16): lane holds column (lane&15); K chunks at
// base=(lane>>4)*8 : [base..base+7] and [base+16..base+23]. Caller passes pointer to base.
DEV bf16x16 make_b(const __bf16* p) {
  bf16x8 c0 = *(const bf16x8*)p;
  bf16x8 c1 = *(const bf16x8*)(p + 16);
  bf16x16 r;
#pragma unroll
  for (int i = 0; i < 8; ++i) { r[i] = c0[i]; r[i + 8] = c1[i]; }
  return r;
}

DEV f32x8 zero8() { f32x8 z = {0.f,0.f,0.f,0.f,0.f,0.f,0.f,0.f}; return z; }

DEV float gelu_f(float x) { return 0.5f * x * (1.0f + erff(x * 0.70710678118654752f)); }

// Async copy 16 bytes global -> LDS (tracked by ASYNCcnt).
DEV void async_b128(unsigned ldsdst, const __bf16* src) {
  asm volatile("global_load_async_to_lds_b128 %0, %1, off"
               :: "v"(ldsdst), "v"((unsigned long long)(uintptr_t)src)
               : "memory");
}

DEV void wait_async0() { asm volatile("s_wait_asynccnt 0x0" ::: "memory"); }

// NT GEMM core with double-buffered async global->LDS staging.
// Block tile 64x128 (8 waves, each 32x32 = 2x2 WMMA tiles), K step 32.
// A (MxK) and W (NxK) row-major bf16; rows clamped during prefetch.
DEV void gemm_lds_core(const __bf16* __restrict__ A, const __bf16* __restrict__ W,
                       int M, int Ncols, int K, int m0b, int n0b, int tid,
                       __bf16* sA, __bf16* sB, f32x8 acc[2][2]) {
  const int lane = tid & 31, wid = tid >> 5;
  const int am = lane & 15, half = lane >> 4;
  const int row = tid >> 2, ch = tid & 3;   // 256 threads -> 64 rows x 4 chunks

  const __bf16* aSrc  = A + (size_t)min(m0b + row, M - 1) * K + ch * 8;
  const __bf16* bSrc0 = W + (size_t)min(n0b + row, Ncols - 1) * K + ch * 8;
  const __bf16* bSrc1 = W + (size_t)min(n0b + 64 + row, Ncols - 1) * K + ch * 8;
  const unsigned aDst  = (unsigned)(uintptr_t)(sA + row * LDA + ch * 8);
  const unsigned bDst0 = (unsigned)(uintptr_t)(sB + row * LDA + ch * 8);
  const unsigned bDst1 = bDst0 + (unsigned)(64 * LDA * 2);

  const __bf16* paBase = sA + ((wid & 1) * 32 + am) * LDA + half * 8;
  const __bf16* pbBase = sB + ((wid >> 1) * 32 + am) * LDA + half * 8;

  const int KT = K / 32;
  // prime buffer 0
  async_b128(aDst, aSrc);
  async_b128(bDst0, bSrc0);
  async_b128(bDst1, bSrc1);
  wait_async0();
  __syncthreads();

  for (int kt = 0; kt < KT; ++kt) {
    const int buf = kt & 1;
    if (kt + 1 < KT) {
      const int nb = 1 - buf;
      const int ko = (kt + 1) * 32;
      async_b128(aDst  + nb * (ABUF * 2), aSrc + ko);
      async_b128(bDst0 + nb * (BBUF * 2), bSrc0 + ko);
      async_b128(bDst1 + nb * (BBUF * 2), bSrc1 + ko);
    }
    const __bf16* pa = paBase + buf * ABUF;
    const __bf16* pb = pbBase + buf * BBUF;
    bf16x16 a0 = make_a(pa);
    bf16x16 a1 = make_a(pa + 16 * LDA);
    bf16x16 b0 = make_b(pb);
    bf16x16 b1 = make_b(pb + 16 * LDA);
    acc[0][0] = wmma_bf16(a0, b0, acc[0][0]);
    acc[0][1] = wmma_bf16(a0, b1, acc[0][1]);
    acc[1][0] = wmma_bf16(a1, b0, acc[1][0]);
    acc[1][1] = wmma_bf16(a1, b1, acc[1][1]);
    wait_async0();       // own wave's prefetch landed
    __syncthreads();     // everyone's landed; readers done before overwrite
  }
}

// ---------------------------------------------------------------- tiny utils

__global__ void cvt_bf16_kernel(const float* __restrict__ in, __bf16* __restrict__ out, int n) {
  int i = blockIdx.x * blockDim.x + threadIdx.x;
  if (i < n) out[i] = (__bf16)in[i];
}

__global__ void zero_kernel(float4* __restrict__ p, size_t n16) {
  size_t i = (size_t)blockIdx.x * blockDim.x + threadIdx.x;
  if (i < n16) p[i] = make_float4(0.f, 0.f, 0.f, 0.f);
}

// ---------------------------------------------------------------- LayerNorms

__global__ __launch_bounds__(384) void ln1_kernel(const float* __restrict__ x,
                                                  const float* __restrict__ g,
                                                  const float* __restrict__ be,
                                                  __bf16* __restrict__ out) {
  int row = blockIdx.x, t = threadIdx.x;
  int lane = t & 31, wid = t >> 5;
  __shared__ float ws[12];
  __shared__ float stat[2];
  float v = x[(size_t)row * C_ + t];
  float s = v;
#pragma unroll
  for (int m = 1; m < 32; m <<= 1) s += __shfl_xor(s, m);
  if (lane == 0) ws[wid] = s;
  __syncthreads();
  if (t == 0) { float tt = 0.f; for (int i = 0; i < 12; ++i) tt += ws[i]; stat[0] = tt / C_; }
  __syncthreads();
  float mu = stat[0];
  float d = v - mu;
  float s2 = d * d;
#pragma unroll
  for (int m = 1; m < 32; m <<= 1) s2 += __shfl_xor(s2, m);
  if (lane == 0) ws[wid] = s2;
  __syncthreads();
  if (t == 0) { float tt = 0.f; for (int i = 0; i < 12; ++i) tt += ws[i]; stat[1] = rsqrtf(tt / C_ + 1e-5f); }
  __syncthreads();
  out[(size_t)row * C_ + t] = (__bf16)(d * stat[1] * g[t] + be[t]);
}

__global__ __launch_bounds__(384) void ln2_kernel(const float* __restrict__ x2,
                                                  const float* __restrict__ g,
                                                  const float* __restrict__ be,
                                                  float* __restrict__ cls,
                                                  __bf16* __restrict__ tok) {
  int row = blockIdx.x, t = threadIdx.x;
  int lane = t & 31, wid = t >> 5;
  __shared__ float ws[12];
  __shared__ float stat[2];
  float v = x2[(size_t)row * C_ + t];
  float s = v;
#pragma unroll
  for (int m = 1; m < 32; m <<= 1) s += __shfl_xor(s, m);
  if (lane == 0) ws[wid] = s;
  __syncthreads();
  if (t == 0) { float tt = 0.f; for (int i = 0; i < 12; ++i) tt += ws[i]; stat[0] = tt / C_; }
  __syncthreads();
  float mu = stat[0];
  float d = v - mu;
  float s2 = d * d;
#pragma unroll
  for (int m = 1; m < 32; m <<= 1) s2 += __shfl_xor(s2, m);
  if (lane == 0) ws[wid] = s2;
  __syncthreads();
  if (t == 0) { float tt = 0.f; for (int i = 0; i < 12; ++i) tt += ws[i]; stat[1] = rsqrtf(tt / C_ + 1e-5f); }
  __syncthreads();
  float y = d * stat[1] * g[t] + be[t];
  int b = row / SEQ, n = row % SEQ;
  if (n == 0) cls[b * C_ + t] = y;
  else        tok[((size_t)b * PIX + (n - 1)) * C_ + t] = (__bf16)y;
}

// ---------------------------------------------------------------- GEMM kernels

// QKV: h(16400x384) * qkv_w(1152x384)^T ; scatter into q/k (B,H,NP,64) and vT (B,H,64,NP)
__global__ __launch_bounds__(256) void qkv_gemm(const __bf16* __restrict__ h,
                                                const __bf16* __restrict__ wq,
                                                __bf16* __restrict__ q,
                                                __bf16* __restrict__ k,
                                                __bf16* __restrict__ vT) {
  __shared__ __align__(16) __bf16 sA[2 * ABUF];
  __shared__ __align__(16) __bf16 sB[2 * BBUF];
  int tid = threadIdx.x, lane = tid & 31, wid = tid >> 5;
  int m0b = blockIdx.y * 64, n0b = blockIdx.x * 128;
  f32x8 acc[2][2] = {zero8(), zero8(), zero8(), zero8()};
  gemm_lds_core(h, wq, MROWS, 3 * C_, C_, m0b, n0b, tid, sA, sB, acc);
  int m0 = m0b + (wid & 1) * 32, n0 = n0b + (wid >> 1) * 32;
  int am = lane & 15, half = lane >> 4;
#pragma unroll
  for (int ti = 0; ti < 2; ++ti)
#pragma unroll
    for (int tj = 0; tj < 2; ++tj)
#pragma unroll
      for (int r = 0; r < 8; ++r) {
        int m = m0 + ti * 16 + half * 8 + r;
        if (m >= MROWS) continue;
        int col = n0 + tj * 16 + am;
        int b = m / SEQ, nrow = m % SEQ;
        int which = col / C_;
        int rem = col % C_;
        int hh = rem / HD, d = rem % HD;
        float v = acc[ti][tj][r];
        size_t bh = (size_t)(b * NH + hh);
        if (which == 0)      q[(bh * NP + nrow) * HD + d] = (__bf16)(v * 0.125f);
        else if (which == 1) k[(bh * NP + nrow) * HD + d] = (__bf16)v;
        else                 vT[(bh * HD + d) * NP + nrow] = (__bf16)v;
      }
}

// proj: o(16400x384) * proj_w(384x384)^T + proj_b + x  ->  x2 (fp32)
__global__ __launch_bounds__(256) void proj_gemm(const __bf16* __restrict__ o,
                                                 const __bf16* __restrict__ wp,
                                                 const float* __restrict__ pb,
                                                 const float* __restrict__ x,
                                                 float* __restrict__ x2) {
  __shared__ __align__(16) __bf16 sA[2 * ABUF];
  __shared__ __align__(16) __bf16 sB[2 * BBUF];
  int tid = threadIdx.x, lane = tid & 31, wid = tid >> 5;
  int m0b = blockIdx.y * 64, n0b = blockIdx.x * 128;
  f32x8 acc[2][2] = {zero8(), zero8(), zero8(), zero8()};
  gemm_lds_core(o, wp, MROWS, C_, C_, m0b, n0b, tid, sA, sB, acc);
  int m0 = m0b + (wid & 1) * 32, n0 = n0b + (wid >> 1) * 32;
  int am = lane & 15, half = lane >> 4;
#pragma unroll
  for (int ti = 0; ti < 2; ++ti)
#pragma unroll
    for (int tj = 0; tj < 2; ++tj)
#pragma unroll
      for (int r = 0; r < 8; ++r) {
        int m = m0 + ti * 16 + half * 8 + r;
        if (m >= MROWS) continue;
        int col = n0 + tj * 16 + am;
        size_t idx = (size_t)m * C_ + col;
        x2[idx] = acc[ti][tj][r] + pb[col] + x[idx];
      }
}

// conv1 (1x1): tok(16384x384) * w1(1536x384)^T ; +b, BN affine, GELU -> y1 fp32
__global__ __launch_bounds__(256) void conv1_gemm(const __bf16* __restrict__ tok,
                                                  const __bf16* __restrict__ w1,
                                                  const float* __restrict__ cb,
                                                  const float* __restrict__ bs,
                                                  const float* __restrict__ bb,
                                                  float* __restrict__ y1) {
  __shared__ __align__(16) __bf16 sA[2 * ABUF];
  __shared__ __align__(16) __bf16 sB[2 * BBUF];
  int tid = threadIdx.x, lane = tid & 31, wid = tid >> 5;
  int m0b = blockIdx.y * 64, n0b = blockIdx.x * 128;
  f32x8 acc[2][2] = {zero8(), zero8(), zero8(), zero8()};
  gemm_lds_core(tok, w1, MTOK, HID, C_, m0b, n0b, tid, sA, sB, acc);
  int m0 = m0b + (wid & 1) * 32, n0 = n0b + (wid >> 1) * 32;
  int am = lane & 15, half = lane >> 4;
#pragma unroll
  for (int ti = 0; ti < 2; ++ti)
#pragma unroll
    for (int tj = 0; tj < 2; ++tj)
#pragma unroll
      for (int r = 0; r < 8; ++r) {
        int m = m0 + ti * 16 + half * 8 + r;
        if (m >= MTOK) continue;
        int col = n0 + tj * 16 + am;
        float v = (acc[ti][tj][r] + cb[col]) * bs[col] + bb[col];
        y1[(size_t)m * HID + col] = gelu_f(v);
      }
}

// conv3 (1x1): y2(16384x1536) * w3(384x1536)^T ; +b, BN affine -> y3 fp32
__global__ __launch_bounds__(256) void conv3_gemm(const __bf16* __restrict__ y2,
                                                  const __bf16* __restrict__ w3,
                                                  const float* __restrict__ cb,
                                                  const float* __restrict__ bs,
                                                  const float* __restrict__ bb,
                                                  float* __restrict__ y3) {
  __shared__ __align__(16) __bf16 sA[2 * ABUF];
  __shared__ __align__(16) __bf16 sB[2 * BBUF];
  int tid = threadIdx.x, lane = tid & 31, wid = tid >> 5;
  int m0b = blockIdx.y * 64, n0b = blockIdx.x * 128;
  f32x8 acc[2][2] = {zero8(), zero8(), zero8(), zero8()};
  gemm_lds_core(y2, w3, MTOK, C_, HID, m0b, n0b, tid, sA, sB, acc);
  int m0 = m0b + (wid & 1) * 32, n0 = n0b + (wid >> 1) * 32;
  int am = lane & 15, half = lane >> 4;
#pragma unroll
  for (int ti = 0; ti < 2; ++ti)
#pragma unroll
    for (int tj = 0; tj < 2; ++tj)
#pragma unroll
      for (int r = 0; r < 8; ++r) {
        int m = m0 + ti * 16 + half * 8 + r;
        if (m >= MTOK) continue;
        int col = n0 + tj * 16 + am;
        y3[(size_t)m * C_ + col] = (acc[ti][tj][r] + cb[col]) * bs[col] + bb[col];
      }
}

// ---------------------------------------------------------------- attention

// Flash attention: one wave = 16 queries; 8 waves/block; key chunks of 32.
// q pre-scaled by hd^-0.5; K (B,H,NP,64), Vt (B,H,64,NP); pads zeroed/masked.
__global__ __launch_bounds__(256) void attn_kernel(const __bf16* __restrict__ q,
                                                   const __bf16* __restrict__ kk,
                                                   const __bf16* __restrict__ vT,
                                                   __bf16* __restrict__ o) {
  __shared__ __align__(16) __bf16 Plds[8][16 * 32];
  int wid = threadIdx.x >> 5, lane = threadIdx.x & 31;
  int bh = blockIdx.y;
  int b = bh / NH, h = bh % NH;
  int q0 = blockIdx.x * 128 + wid * 16;
  if (q0 >= SEQ) return;
  int am = lane & 15, half = lane >> 4;

  const __bf16* Q  = q  + ((size_t)bh * NP + q0) * HD;
  const __bf16* Km = kk + (size_t)bh * NP * HD;
  const __bf16* Vt = vT + (size_t)bh * HD * NP;

  const __bf16* qrow = Q + (size_t)am * HD + half * 8;
  bf16x16 aq0 = make_a(qrow);        // d = 0..31
  bf16x16 aq1 = make_a(qrow + 32);   // d = 32..63

  f32x8 oacc[4] = {zero8(), zero8(), zero8(), zero8()};
  float mrow[8], lrow[8];
#pragma unroll
  for (int r = 0; r < 8; ++r) { mrow[r] = -1e30f; lrow[r] = 0.f; }

  for (int kc = 0; kc < NP; kc += 32) {
    // scores S = q @ k^T for 32-key chunk (two 16-col tiles)
    const __bf16* kb = Km + (size_t)(kc + am) * HD + half * 8;
    f32x8 s0 = zero8(), s1 = zero8();
    s0 = wmma_bf16(aq0, make_b(kb), s0);
    s0 = wmma_bf16(aq1, make_b(kb + 32), s0);
    const __bf16* kb1 = kb + (size_t)16 * HD;
    s1 = wmma_bf16(aq0, make_b(kb1), s1);
    s1 = wmma_bf16(aq1, make_b(kb1 + 32), s1);

    bool ok0 = (kc + am) < SEQ;
    bool ok1 = (kc + 16 + am) < SEQ;
#pragma unroll
    for (int r = 0; r < 8; ++r) {
      float v0 = ok0 ? s0[r] : -1e30f;
      float v1 = ok1 ? s1[r] : -1e30f;
      float mx = fmaxf(v0, v1);
#pragma unroll
      for (int msk = 1; msk < 16; msk <<= 1) mx = fmaxf(mx, __shfl_xor(mx, msk));
      float mnew = fmaxf(mrow[r], mx);
      float alpha = __expf(mrow[r] - mnew);
      float p0 = __expf(v0 - mnew);
      float p1 = __expf(v1 - mnew);
      float rs = p0 + p1;
#pragma unroll
      for (int msk = 1; msk < 16; msk <<= 1) rs += __shfl_xor(rs, msk);
      lrow[r] = lrow[r] * alpha + rs;
      mrow[r] = mnew;
      oacc[0][r] *= alpha; oacc[1][r] *= alpha;
      oacc[2][r] *= alpha; oacc[3][r] *= alpha;
      int prow = r + half * 8;
      Plds[wid][prow * 32 + am]      = (__bf16)p0;
      Plds[wid][prow * 32 + 16 + am] = (__bf16)p1;
    }
    // reload P (16x32) as A fragment; same wave, LDS is in-order
    const __bf16* pp = &Plds[wid][(size_t)am * 32 + half * 8];
    bf16x16 ap = make_a(pp);
    // V B-fragments: column d = tj*16 + am, keys kc + half*8 + {0..7, 16..23}
    const __bf16* vb = Vt + (size_t)am * NP + kc + half * 8;
    oacc[0] = wmma_bf16(ap, make_b(vb), oacc[0]);
    oacc[1] = wmma_bf16(ap, make_b(vb + (size_t)16 * NP), oacc[1]);
    oacc[2] = wmma_bf16(ap, make_b(vb + (size_t)32 * NP), oacc[2]);
    oacc[3] = wmma_bf16(ap, make_b(vb + (size_t)48 * NP), oacc[3]);
  }

#pragma unroll
  for (int tj = 0; tj < 4; ++tj)
#pragma unroll
    for (int r = 0; r < 8; ++r) {
      int row = q0 + half * 8 + r;
      if (row < SEQ) {
        int col = h * HD + tj * 16 + am;
        o[((size_t)b * SEQ + row) * C_ + col] = (__bf16)(oacc[tj][r] / lrow[r]);
      }
    }
}

// ---------------------------------------------------------------- depthwise

__global__ __launch_bounds__(256) void dwconv_kernel(const float* __restrict__ y1,
                                                     const float* __restrict__ w2,
                                                     const float* __restrict__ cb,
                                                     const float* __restrict__ bs,
                                                     const float* __restrict__ bb,
                                                     __bf16* __restrict__ y2) {
  int ch = blockIdx.x * 256 + threadIdx.x;
  int p  = blockIdx.y;
  int b  = blockIdx.z;
  int r = p >> 5, c = p & 31;
  const float* base = y1 + (size_t)b * PIX * HID + ch;
  const float* wp = w2 + (size_t)ch * 9;
  float acc = 0.f;
#pragma unroll
  for (int i = 0; i < 3; ++i) {
    int rr = r + i - 1;
    if ((unsigned)rr >= 32u) continue;
#pragma unroll
    for (int j = 0; j < 3; ++j) {
      int cc = c + j - 1;
      if ((unsigned)cc >= 32u) continue;
      acc += base[(size_t)(rr * 32 + cc) * HID] * wp[i * 3 + j];
    }
  }
  float v = (acc + cb[ch]) * bs[ch] + bb[ch];
  float shortcut = base[(size_t)p * HID];
  y2[((size_t)b * PIX + p) * HID + ch] = (__bf16)(shortcut + gelu_f(v));
}

// ---------------------------------------------------------------- SE branch

__global__ __launch_bounds__(384) void pool_kernel(const float* __restrict__ y3,
                                                   float* __restrict__ pool) {
  int b = blockIdx.x, c = threadIdx.x;
  const float* p = y3 + (size_t)b * PIX * C_ + c;
  float s = 0.f;
  for (int i = 0; i < PIX; ++i) s += p[(size_t)i * C_];
  pool[b * C_ + c] = s * (1.0f / PIX);
}

__global__ __launch_bounds__(128) void se_kernel(const float* __restrict__ pool,
                                                 const float* __restrict__ cw,
                                                 const float* __restrict__ cb,
                                                 const float* __restrict__ ew,
                                                 const float* __restrict__ eb,
                                                 float* __restrict__ sew) {
  __shared__ float pl[C_];
  __shared__ float z[C4];
  int b = blockIdx.x, t = threadIdx.x;
  for (int i = t; i < C_; i += 128) pl[i] = pool[b * C_ + i];
  __syncthreads();
  if (t < C4) {
    float s = cb[t];
    const float* w = cw + (size_t)t * C_;
    for (int i = 0; i < C_; ++i) s += pl[i] * w[i];
    z[t] = gelu_f(s);
  }
  __syncthreads();
  for (int c = t; c < C_; c += 128) {
    float s = eb[c];
    const float* w = ew + (size_t)c * C4;
    for (int j = 0; j < C4; ++j) s += z[j] * w[j];
    sew[b * C_ + c] = s;
  }
}

// ---------------------------------------------------------------- final merge

__global__ __launch_bounds__(384) void final_kernel(const float* __restrict__ x2,
                                                    const float* __restrict__ cls,
                                                    const float* __restrict__ sew,
                                                    const float* __restrict__ y3,
                                                    float* __restrict__ out) {
  int row = blockIdx.x, t = threadIdx.x;
  int b = row / SEQ, n = row % SEQ;
  float base = x2[(size_t)row * C_ + t];
  float add = (n == 0) ? cls[b * C_ + t] * sew[b * C_ + t]
                       : y3[((size_t)b * PIX + (n - 1)) * C_ + t];
  out[(size_t)row * C_ + t] = base + add;
}

// ---------------------------------------------------------------- launch

extern "C" void kernel_launch(void* const* d_in, const int* in_sizes, int n_in,
                              void* d_out, int out_size, void* d_ws, size_t ws_size,
                              hipStream_t stream) {
  (void)in_sizes; (void)n_in; (void)out_size; (void)ws_size;
  const float* x       = (const float*)d_in[0];
  const float* ln1_g   = (const float*)d_in[1];
  const float* ln1_b   = (const float*)d_in[2];
  const float* qkv_w   = (const float*)d_in[3];
  const float* proj_w  = (const float*)d_in[4];
  const float* proj_b  = (const float*)d_in[5];
  const float* ln2_g   = (const float*)d_in[6];
  const float* ln2_b   = (const float*)d_in[7];
  const float* conv1_w = (const float*)d_in[8];
  const float* conv1_b = (const float*)d_in[9];
  const float* conv2_w = (const float*)d_in[10];
  const float* conv2_b = (const float*)d_in[11];
  const float* conv3_w = (const float*)d_in[12];
  const float* conv3_b = (const float*)d_in[13];
  const float* bn1_s   = (const float*)d_in[14];
  const float* bn1_b   = (const float*)d_in[15];
  const float* bn2_s   = (const float*)d_in[16];
  const float* bn2_b   = (const float*)d_in[17];
  const float* bn3_s   = (const float*)d_in[18];
  const float* bn3_b   = (const float*)d_in[19];
  const float* comp_w  = (const float*)d_in[20];
  const float* comp_b  = (const float*)d_in[21];
  const float* exc_w   = (const float*)d_in[22];
  const float* exc_b   = (const float*)d_in[23];

  char* base = (char*)d_ws;
  size_t off = 0;
  auto take = [&](size_t bytes) -> void* {
    void* p = base + off;
    off = (off + bytes + 255) & ~(size_t)255;
    return p;
  };

  __bf16* wq   = (__bf16*)take((size_t)3 * C_ * C_ * 2);
  __bf16* wp   = (__bf16*)take((size_t)C_ * C_ * 2);
  __bf16* w1   = (__bf16*)take((size_t)HID * C_ * 2);
  __bf16* w3   = (__bf16*)take((size_t)C_ * HID * 2);
  __bf16* hbf  = (__bf16*)take((size_t)MROWS * C_ * 2);
  __bf16* qb   = (__bf16*)take((size_t)B_ * NH * NP * HD * 2);
  __bf16* kb   = (__bf16*)take((size_t)B_ * NH * NP * HD * 2);
  __bf16* vT   = (__bf16*)take((size_t)B_ * NH * HD * NP * 2);
  __bf16* ob   = (__bf16*)take((size_t)MROWS * C_ * 2);
  float*  x2   = (float*)take((size_t)MROWS * C_ * 4);
  float*  cls  = (float*)take((size_t)B_ * C_ * 4);
  __bf16* tok  = (__bf16*)take((size_t)MTOK * C_ * 2);
  float*  y1   = (float*)take((size_t)MTOK * HID * 4);
  __bf16* y2   = (__bf16*)take((size_t)MTOK * HID * 2);
  float*  y3   = (float*)take((size_t)MTOK * C_ * 4);
  float*  pool = (float*)take((size_t)B_ * C_ * 4);
  float*  sew  = (float*)take((size_t)B_ * C_ * 4);

  // weight conversions fp32 -> bf16
  { int n = 3 * C_ * C_; cvt_bf16_kernel<<<(n + 255) / 256, 256, 0, stream>>>(qkv_w, wq, n); }
  { int n = C_ * C_;     cvt_bf16_kernel<<<(n + 255) / 256, 256, 0, stream>>>(proj_w, wp, n); }
  { int n = HID * C_;    cvt_bf16_kernel<<<(n + 255) / 256, 256, 0, stream>>>(conv1_w, w1, n); }
  { int n = C_ * HID;    cvt_bf16_kernel<<<(n + 255) / 256, 256, 0, stream>>>(conv3_w, w3, n); }

  // zero padded q/k/vT each call (deterministic; ws is poisoned once)
  {
    size_t n16 = (size_t)B_ * NH * NP * HD * 2 / 16;
    int g = (int)((n16 + 255) / 256);
    zero_kernel<<<g, 256, 0, stream>>>((float4*)qb, n16);
    zero_kernel<<<g, 256, 0, stream>>>((float4*)kb, n16);
    zero_kernel<<<g, 256, 0, stream>>>((float4*)vT, n16);
  }

  ln1_kernel<<<MROWS, 384, 0, stream>>>(x, ln1_g, ln1_b, hbf);
  qkv_gemm<<<dim3(9, (MROWS + 63) / 64), 256, 0, stream>>>(hbf, wq, qb, kb, vT);
  attn_kernel<<<dim3(9, B_ * NH), 256, 0, stream>>>(qb, kb, vT, ob);
  proj_gemm<<<dim3(3, (MROWS + 63) / 64), 256, 0, stream>>>(ob, wp, proj_b, x, x2);
  ln2_kernel<<<MROWS, 384, 0, stream>>>(x2, ln2_g, ln2_b, cls, tok);
  conv1_gemm<<<dim3(12, MTOK / 64), 256, 0, stream>>>(tok, w1, conv1_b, bn1_s, bn1_b, y1);
  dwconv_kernel<<<dim3(HID / 256, PIX, B_), 256, 0, stream>>>(y1, conv2_w, conv2_b, bn2_s, bn2_b, y2);
  conv3_gemm<<<dim3(3, MTOK / 64), 256, 0, stream>>>(y2, w3, conv3_b, bn3_s, bn3_b, y3);
  pool_kernel<<<B_, 384, 0, stream>>>(y3, pool);
  se_kernel<<<B_, 128, 0, stream>>>(pool, comp_w, comp_b, exc_w, exc_b, sew);
  final_kernel<<<MROWS, 384, 0, stream>>>(x2, cls, sew, y3, (float*)d_out);
}
